// MPNNet_Parametric_42597485642062
// MI455X (gfx1250) — matrix operand order repaired and checked by reading, not compiled, not run on hardware
//
#include <hip/hip_runtime.h>
#include <hip/hip_bf16.h>
#include <utility>

// ---------------- problem constants ----------------
#define DIM       64
#define NUM_FEAT  14
#define EDGE_FEAT 4
#define N_NODES   12800
#define N_EDGES   25600
#define N_GRAPHS  256

typedef __attribute__((ext_vector_type(16))) _Float16 v16h;
typedef __attribute__((ext_vector_type(8)))  _Float16 v8h;
typedef __attribute__((ext_vector_type(2)))  _Float16 v2h;
typedef __attribute__((ext_vector_type(8)))  float    v8f;

// ---------------- small device helpers ----------------
__device__ __forceinline__ float lrelu(float v) { return v > 0.f ? v : 0.01f * v; }
__device__ __forceinline__ float sigm(float x) { return 1.f / (1.f + __expf(-x)); }
__device__ __forceinline__ float ftanh(float x) {
  x = fminf(15.f, fmaxf(-15.f, x));
  float e = __expf(2.f * x);
  return (e - 1.f) / (e + 1.f);
}
__device__ __forceinline__ void atomAddF(float* p, float v) { unsafeAtomicAdd(p, v); }
__device__ __forceinline__ void atomMaxF(float* addr, float val) {
  unsigned int* ua = (unsigned int*)addr;
  unsigned int old = __hip_atomic_load(ua, __ATOMIC_RELAXED, __HIP_MEMORY_SCOPE_AGENT);
  while (__uint_as_float(old) < val) {
    unsigned int assumed = old;
    old = atomicCAS(ua, assumed, __float_as_uint(val));
    if (old == assumed) break;
  }
}

// CDNA5 WMMA f32 = f16 x f16 + f32, 16x16x32
__device__ __forceinline__ v8f wmma16(v16h a, v16h b, v8f c) {
  return __builtin_amdgcn_wmma_f32_16x16x32_f16(false, a, false, b, (short)0, c, false, false);
}

// A fragment (16x32 f16) gather from row-major [row][64] f16 storage.
// lane<16: K = {0..7, 16..23}; lane>=16: K = {8..15, 24..31} (+kofs)
__device__ __forceinline__ v16h make_afrag(const _Float16* rowp, int lane, int kofs) {
  int kb = (lane & 16) ? 8 : 0;
  v8h lo = *(const v8h*)(rowp + kofs + kb);
  v8h hi = *(const v8h*)(rowp + kofs + kb + 16);
  v16h r;
#pragma unroll
  for (int j = 0; j < 8; ++j) { r[j] = lo[j]; r[8 + j] = hi[j]; }
  return r;
}
// B fragment: packed contiguously per (tile, lane): 16 halves each
__device__ __forceinline__ v16h load_bfrag(const _Float16* packed, int tile, int lane) {
  return *(const v16h*)(packed + ((size_t)tile * 32 + lane) * 16);
}

// ---------------- weight packing into B-fragment layout ----------------
// B tile (kt,nt) of a K x N matrix, stored as [tile][lane][j] halves where
// K = kt*32 + (lane<16?0:8) + (j&7) + (j>=8?16:0), n = nt*16 + lane%16.
// transpose==1: B[k][n] = W[n*K + k]  (for "@ W.T" layers, W is [N][K])
// transpose==0: B[k][n] = W[k*N + n]
__global__ void mpnn_pack_b(const float* __restrict__ W, _Float16* __restrict__ dst,
                            int N, int K, int transpose, int total) {
  int idx = blockIdx.x * blockDim.x + threadIdx.x;
  if (idx >= total) return;
  int NT = N >> 4;
  int t = idx >> 9, r = idx & 511;
  int lane = r >> 4, j = r & 15;
  int kt = t / NT, nt = t % NT;
  int k = kt * 32 + ((lane < 16) ? 0 : 8) + (j & 7) + ((j & 8) ? 16 : 0);
  int n = nt * 16 + (lane & 15);
  float v = transpose ? W[(size_t)n * K + k] : W[(size_t)k * N + n];
  dst[idx] = (_Float16)v;
}

// ---------------- misc elementwise kernels ----------------
__global__ void mpnn_zero_f32(float* __restrict__ p, int n) {
  int i = blockIdx.x * blockDim.x + threadIdx.x;
  if (i < n) p[i] = 0.f;
}

// out = lrelu(x @ lin0_w.T + b), store f32 + f16 copies (initial state)
__global__ void mpnn_lin0(const float* __restrict__ x, const float* __restrict__ w,
                          const float* __restrict__ b, float* __restrict__ s32,
                          _Float16* __restrict__ s16) {
  int idx = blockIdx.x * blockDim.x + threadIdx.x;
  if (idx >= N_NODES * DIM) return;
  int node = idx >> 6, col = idx & 63;
  const float* xr = x + node * NUM_FEAT;
  const float* wr = w + col * NUM_FEAT;
  float acc = b[col];
#pragma unroll
  for (int k = 0; k < NUM_FEAT; ++k) acc += xr[k] * wr[k];
  acc = lrelu(acc);
  s32[idx] = acc;
  s16[idx] = (_Float16)acc;
}

// F = lrelu(edge_attr @ en1_w.T + en1_b), f16 row-major [E][64]
__global__ void mpnn_edge_hidden(const float* __restrict__ ea, const float* __restrict__ w1,
                                 const float* __restrict__ b1, _Float16* __restrict__ F) {
  int idx = blockIdx.x * blockDim.x + threadIdx.x;
  if (idx >= N_EDGES * DIM) return;
  int e = idx >> 6, col = idx & 63;
  const float* er = ea + e * EDGE_FEAT;
  const float* wr = w1 + col * EDGE_FEAT;
  float acc = b1[col] + er[0] * wr[0] + er[1] * wr[1] + er[2] * wr[2] + er[3] * wr[3];
  F[idx] = (_Float16)lrelu(acc);
}

__global__ void mpnn_deg_count(const int* __restrict__ ei, float* __restrict__ deg) {
  int e = blockIdx.x * blockDim.x + threadIdx.x;
  if (e >= N_EDGES) return;
  atomAddF(&deg[ei[N_EDGES + e]], 1.0f);
}
__global__ void mpnn_deg_recip(float* __restrict__ deg) {
  int i = blockIdx.x * blockDim.x + threadIdx.x;
  if (i < N_NODES) deg[i] = 1.0f / fmaxf(deg[i], 1.0f);
}

// ---------------- big WMMA GEMM: w_e = F @ en2_w.T + en2_b (f16 out) ----------------
// M = 25600 edges, N = 4096, K = 64. One wave per 16x64 output strip:
// A fragment loaded once, reused across 4 N-tiles (8 WMMAs per wave).
// w_e is streamed (210 MB > 192 MB L2) -> non-temporal stores.
__global__ void mpnn_edge_gemm(const _Float16* __restrict__ F, const _Float16* __restrict__ en2B,
                               const float* __restrict__ en2b, _Float16* __restrict__ W) {
  const int NT = 4096 / 16;  // 256 column tiles
  int wave = (blockIdx.x * blockDim.x + threadIdx.x) >> 5;
  int lane = threadIdx.x & 31;
  if (wave >= (N_EDGES / 16) * (NT / 4)) return;
  int mt = wave >> 6;             // edge tile
  int ntg = (wave & 63) << 2;     // first of 4 consecutive column tiles
  const _Float16* arow = F + (size_t)(mt * 16 + (lane & 15)) * DIM;
  v16h a0 = make_afrag(arow, lane, 0);
  v16h a1 = make_afrag(arow, lane, 32);
  int row0 = mt * 16 + ((lane < 16) ? 0 : 8);
#pragma unroll
  for (int q = 0; q < 4; ++q) {
    int nt = ntg + q;
    v8f c = {};
    c = wmma16(a0, load_bfrag(en2B, nt, lane), c);
    c = wmma16(a1, load_bfrag(en2B, NT + nt, lane), c);
    int col = nt * 16 + (lane & 15);
    float bias = en2b[col];
#pragma unroll
    for (int v = 0; v < 8; ++v)
      __builtin_nontemporal_store((_Float16)(c[v] + bias),
                                  &W[(size_t)(row0 + v) * 4096 + col]);
  }
}

// ---------------- per-edge matvec + scatter (bandwidth-bound -> VALU) ----------------
// One wave per edge: msg[o] = sum_i state[src][i] * w_e[e][i][o]; atomics into agg[dst].
// w_e reads are non-temporal: 210 MB/pass must not evict the hot L2 state.
__global__ void mpnn_msg(const float* __restrict__ state, const _Float16* __restrict__ W,
                         const int* __restrict__ ei, float* __restrict__ agg) {
  int e = (blockIdx.x * blockDim.x + threadIdx.x) >> 5;
  int lane = threadIdx.x & 31;
  if (e >= N_EDGES) return;
  int src = ei[e], dst = ei[N_EDGES + e];
  const float* srow = state + (size_t)src * DIM;
  float r0 = srow[lane], r1 = srow[lane + 32];
  const _Float16* wrow = W + (size_t)e * 4096;
  float acc0 = 0.f, acc1 = 0.f;
#pragma unroll 8
  for (int i = 0; i < 32; ++i) {
    float s = __int_as_float(__builtin_amdgcn_readlane(__float_as_int(r0), i));
    v2h w = __builtin_nontemporal_load((const v2h*)(wrow + i * DIM + 2 * lane));
    acc0 += s * (float)w[0];
    acc1 += s * (float)w[1];
  }
#pragma unroll 8
  for (int i = 0; i < 32; ++i) {
    float s = __int_as_float(__builtin_amdgcn_readlane(__float_as_int(r1), i));
    v2h w = __builtin_nontemporal_load((const v2h*)(wrow + (i + 32) * DIM + 2 * lane));
    acc0 += s * (float)w[0];
    acc1 += s * (float)w[1];
  }
  float* arow = agg + (size_t)dst * DIM + 2 * lane;
  atomAddF(arow, acc0);
  atomAddF(arow + 1, acc1);
}

// ---------------- conv: m = lrelu(agg*rdeg + state@root_w + conv_b) (WMMA) ----------
// One wave per 16-node tile: A loaded once, all 4 column tiles (8 WMMAs).
__global__ void mpnn_conv(const _Float16* __restrict__ s16, const _Float16* __restrict__ rootB,
                          const float* __restrict__ convb, const float* __restrict__ agg,
                          const float* __restrict__ rdeg, _Float16* __restrict__ m16) {
  int wave = (blockIdx.x * blockDim.x + threadIdx.x) >> 5;
  int lane = threadIdx.x & 31;
  if (wave >= N_NODES / 16) return;
  int mt = wave;
  const _Float16* arow = s16 + (size_t)(mt * 16 + (lane & 15)) * DIM;
  v16h a0 = make_afrag(arow, lane, 0);
  v16h a1 = make_afrag(arow, lane, 32);
  int row0 = mt * 16 + ((lane < 16) ? 0 : 8);
#pragma unroll
  for (int nt = 0; nt < 4; ++nt) {
    v8f c = {};
    c = wmma16(a0, load_bfrag(rootB, 0 * 4 + nt, lane), c);
    c = wmma16(a1, load_bfrag(rootB, 1 * 4 + nt, lane), c);
    int col = nt * 16 + (lane & 15);
    float bias = convb[col];
#pragma unroll
    for (int v = 0; v < 8; ++v) {
      int row = row0 + v;
      float val = c[v] + bias + agg[(size_t)row * DIM + col] * rdeg[row];
      m16[(size_t)row * DIM + col] = (_Float16)lrelu(val);
    }
  }
}

// ---------------- GRU cell (WMMA for gi = m@Wih.T, gh = h@Whh.T) ----------------
__global__ void mpnn_gru(const _Float16* __restrict__ m16, const _Float16* __restrict__ h16,
                         const float* __restrict__ h32, const _Float16* __restrict__ wihB,
                         const _Float16* __restrict__ whhB, const float* __restrict__ bih,
                         const float* __restrict__ bhh, float* __restrict__ o32,
                         _Float16* __restrict__ o16) {
  const int NT = 12;  // 192/16 column tiles of the packed GRU weights
  int wave = (blockIdx.x * blockDim.x + threadIdx.x) >> 5;
  int lane = threadIdx.x & 31;
  if (wave >= (N_NODES / 16) * 4) return;
  int mt = wave >> 2, nt = wave & 3;  // nt: which 16-column slice of d=64
  const _Float16* mrow = m16 + (size_t)(mt * 16 + (lane & 15)) * DIM;
  const _Float16* hrow = h16 + (size_t)(mt * 16 + (lane & 15)) * DIM;
  v16h am0 = make_afrag(mrow, lane, 0), am1 = make_afrag(mrow, lane, 32);
  v16h ah0 = make_afrag(hrow, lane, 0), ah1 = make_afrag(hrow, lane, 32);
  v8f cir = {}, ciz = {}, cin = {}, chr_ = {}, chz = {}, chn = {};
  // gate g columns live at tiles g*4 + nt
  cir = wmma16(am0, load_bfrag(wihB, 0 * NT + 0 * 4 + nt, lane), cir);
  cir = wmma16(am1, load_bfrag(wihB, 1 * NT + 0 * 4 + nt, lane), cir);
  ciz = wmma16(am0, load_bfrag(wihB, 0 * NT + 1 * 4 + nt, lane), ciz);
  ciz = wmma16(am1, load_bfrag(wihB, 1 * NT + 1 * 4 + nt, lane), ciz);
  cin = wmma16(am0, load_bfrag(wihB, 0 * NT + 2 * 4 + nt, lane), cin);
  cin = wmma16(am1, load_bfrag(wihB, 1 * NT + 2 * 4 + nt, lane), cin);
  chr_ = wmma16(ah0, load_bfrag(whhB, 0 * NT + 0 * 4 + nt, lane), chr_);
  chr_ = wmma16(ah1, load_bfrag(whhB, 1 * NT + 0 * 4 + nt, lane), chr_);
  chz = wmma16(ah0, load_bfrag(whhB, 0 * NT + 1 * 4 + nt, lane), chz);
  chz = wmma16(ah1, load_bfrag(whhB, 1 * NT + 1 * 4 + nt, lane), chz);
  chn = wmma16(ah0, load_bfrag(whhB, 0 * NT + 2 * 4 + nt, lane), chn);
  chn = wmma16(ah1, load_bfrag(whhB, 1 * NT + 2 * 4 + nt, lane), chn);

  int col = nt * 16 + (lane & 15);
  float bir = bih[col], biz = bih[64 + col], bin = bih[128 + col];
  float bhr = bhh[col], bhz = bhh[64 + col], bhn = bhh[128 + col];
  int row0 = mt * 16 + ((lane < 16) ? 0 : 8);
#pragma unroll
  for (int v = 0; v < 8; ++v) {
    int row = row0 + v;
    float r = sigm(cir[v] + bir + chr_[v] + bhr);
    float z = sigm(ciz[v] + biz + chz[v] + bhz);
    float nn = ftanh(cin[v] + bin + r * (chn[v] + bhn));
    float hold = h32[(size_t)row * DIM + col];
    float hnew = (1.f - z) * nn + z * hold;
    o32[(size_t)row * DIM + col] = hnew;
    o16[(size_t)row * DIM + col] = (_Float16)hnew;
  }
}

// ---------------- Set2Set (processing_steps=1; q is graph-independent) ----------
__global__ void mpnn_q0(const float* __restrict__ bi, const float* __restrict__ bh,
                        float* __restrict__ q0) {
  int j = threadIdx.x;
  if (j >= DIM) return;
  float ig = sigm(bi[j] + bh[j]);
  float gg = ftanh(bi[128 + j] + bh[128 + j]);
  float og = sigm(bi[192 + j] + bh[192 + j]);
  q0[j] = og * ftanh(ig * gg);  // cl = f*0 + i*g
}
__global__ void mpnn_s2s_init(float* __restrict__ emax, float* __restrict__ denom) {
  int g = blockIdx.x * blockDim.x + threadIdx.x;
  if (g >= N_GRAPHS) return;
  emax[g] = -3.0e38f;
  denom[g] = 0.f;
}
__global__ void mpnn_e(const float* __restrict__ state, const float* __restrict__ q0,
                       const int* __restrict__ batch, float* __restrict__ e,
                       float* __restrict__ emax) {
  int n = blockIdx.x * blockDim.x + threadIdx.x;
  if (n >= N_NODES) return;
  const float* s = state + (size_t)n * DIM;
  float acc = 0.f;
#pragma unroll
  for (int k = 0; k < DIM; ++k) acc += s[k] * q0[k];
  e[n] = acc;
  atomMaxF(&emax[batch[n]], acc);
}
__global__ void mpnn_ee(const float* __restrict__ e, const float* __restrict__ emax,
                        const int* __restrict__ batch, float* __restrict__ ee,
                        float* __restrict__ denom) {
  int n = blockIdx.x * blockDim.x + threadIdx.x;
  if (n >= N_NODES) return;
  int g = batch[n];
  float v = __expf(e[n] - emax[g]);
  ee[n] = v;
  atomAddF(&denom[g], v);
}
__global__ void mpnn_rvec(const float* __restrict__ state, const float* __restrict__ ee,
                          const float* __restrict__ denom, const int* __restrict__ batch,
                          float* __restrict__ rvec) {
  int idx = blockIdx.x * blockDim.x + threadIdx.x;
  if (idx >= N_NODES * DIM) return;
  int n = idx >> 6, k = idx & 63;
  int g = batch[n];
  float a = ee[n] / denom[g];
  atomAddF(&rvec[(size_t)g * DIM + k], a * state[idx]);
}
__global__ void mpnn_out(const float* __restrict__ q0, const float* __restrict__ rvec,
                         float* __restrict__ out) {
  int idx = blockIdx.x * blockDim.x + threadIdx.x;
  if (idx >= N_GRAPHS * 2 * DIM) return;
  int g = idx >> 7, j = idx & 127;
  out[idx] = (j < DIM) ? q0[j] : rvec[(size_t)g * DIM + (j - DIM)];
}

// ---------------- host launcher ----------------
extern "C" void kernel_launch(void* const* d_in, const int* in_sizes, int n_in,
                              void* d_out, int out_size, void* d_ws, size_t ws_size,
                              hipStream_t stream) {
  (void)in_sizes; (void)n_in; (void)out_size; (void)ws_size;
  const float* x       = (const float*)d_in[0];
  const int*   ei      = (const int*)d_in[1];
  const float* ea      = (const float*)d_in[2];
  const int*   batch   = (const int*)d_in[3];
  const float* lin0_w  = (const float*)d_in[4];
  const float* lin0_b  = (const float*)d_in[5];
  const float* en1_w   = (const float*)d_in[6];
  const float* en1_b   = (const float*)d_in[7];
  const float* en2_w   = (const float*)d_in[8];
  const float* en2_b   = (const float*)d_in[9];
  const float* root_w  = (const float*)d_in[10];
  const float* conv_b  = (const float*)d_in[11];
  const float* gru_wih = (const float*)d_in[12];
  const float* gru_whh = (const float*)d_in[13];
  const float* gru_bih = (const float*)d_in[14];
  const float* gru_bhh = (const float*)d_in[15];
  const float* lstm_bi = (const float*)d_in[18];
  const float* lstm_bh = (const float*)d_in[19];

  char* ws = (char*)d_ws;
  size_t off = 0;
  auto alloc = [&](size_t bytes) -> void* {
    void* p = ws + off;
    off = (off + bytes + 255) & ~(size_t)255;
    return p;
  };
  _Float16* w_e   = (_Float16*)alloc((size_t)N_EDGES * 4096 * 2);  // 209.7 MB
  _Float16* en2B  = (_Float16*)alloc(2 * 256 * 512 * 2);
  _Float16* rootB = (_Float16*)alloc(2 * 4 * 512 * 2);
  _Float16* wihB  = (_Float16*)alloc(2 * 12 * 512 * 2);
  _Float16* whhB  = (_Float16*)alloc(2 * 12 * 512 * 2);
  _Float16* F     = (_Float16*)alloc((size_t)N_EDGES * DIM * 2);
  _Float16* s16a  = (_Float16*)alloc((size_t)N_NODES * DIM * 2);
  _Float16* s16b  = (_Float16*)alloc((size_t)N_NODES * DIM * 2);
  _Float16* m16   = (_Float16*)alloc((size_t)N_NODES * DIM * 2);
  float* s32a  = (float*)alloc((size_t)N_NODES * DIM * 4);
  float* s32b  = (float*)alloc((size_t)N_NODES * DIM * 4);
  float* agg   = (float*)alloc((size_t)N_NODES * DIM * 4);
  float* rdeg  = (float*)alloc(N_NODES * 4);
  float* e_buf = (float*)alloc(N_NODES * 4);
  float* ee_buf= (float*)alloc(N_NODES * 4);
  float* q0    = (float*)alloc(DIM * 4);
  float* emax  = (float*)alloc(N_GRAPHS * 4);
  float* denom = (float*)alloc(N_GRAPHS * 4);
  float* rvec  = (float*)alloc((size_t)N_GRAPHS * DIM * 4);

  const int B = 256;
  auto gb = [](int n, int b) { return (n + b - 1) / b; };

  // weight packing (f16 B-fragment layout)
  mpnn_pack_b<<<gb(2 * 256 * 512, B), B, 0, stream>>>(en2_w, en2B, 4096, 64, 1, 2 * 256 * 512);
  mpnn_pack_b<<<gb(2 * 4 * 512, B), B, 0, stream>>>(root_w, rootB, 64, 64, 0, 2 * 4 * 512);
  mpnn_pack_b<<<gb(2 * 12 * 512, B), B, 0, stream>>>(gru_wih, wihB, 192, 64, 1, 2 * 12 * 512);
  mpnn_pack_b<<<gb(2 * 12 * 512, B), B, 0, stream>>>(gru_whh, whhB, 192, 64, 1, 2 * 12 * 512);

  // initial node state + edge features + degrees
  mpnn_lin0<<<gb(N_NODES * DIM, B), B, 0, stream>>>(x, lin0_w, lin0_b, s32a, s16a);
  mpnn_edge_hidden<<<gb(N_EDGES * DIM, B), B, 0, stream>>>(ea, en1_w, en1_b, F);
  mpnn_zero_f32<<<gb(N_NODES, B), B, 0, stream>>>(rdeg, N_NODES);
  mpnn_deg_count<<<gb(N_EDGES, B), B, 0, stream>>>(ei, rdeg);
  mpnn_deg_recip<<<gb(N_NODES, B), B, 0, stream>>>(rdeg);

  // big one-time edge-filter GEMM (WMMA): 1600 x 64 waves, 8 WMMAs each
  mpnn_edge_gemm<<<gb((N_EDGES / 16) * 64 * 32, B), B, 0, stream>>>(F, en2B, en2_b, w_e);

  // 6 message-passing iterations (ping-pong state buffers)
  float* s32_cur = s32a; _Float16* s16_cur = s16a;
  float* s32_nxt = s32b; _Float16* s16_nxt = s16b;
  for (int t = 0; t < 6; ++t) {
    mpnn_zero_f32<<<gb(N_NODES * DIM, B), B, 0, stream>>>(agg, N_NODES * DIM);
    mpnn_msg<<<gb(N_EDGES * 32, B), B, 0, stream>>>(s32_cur, w_e, ei, agg);
    mpnn_conv<<<gb((N_NODES / 16) * 32, B), B, 0, stream>>>(s16_cur, rootB, conv_b, agg, rdeg, m16);
    mpnn_gru<<<gb((N_NODES / 16) * 4 * 32, B), B, 0, stream>>>(m16, s16_cur, s32_cur, wihB, whhB,
                                                               gru_bih, gru_bhh, s32_nxt, s16_nxt);
    std::swap(s32_cur, s32_nxt);
    std::swap(s16_cur, s16_nxt);
  }

  // Set2Set readout
  mpnn_q0<<<1, 64, 0, stream>>>(lstm_bi, lstm_bh, q0);
  mpnn_s2s_init<<<1, 256, 0, stream>>>(emax, denom);
  mpnn_zero_f32<<<gb(N_GRAPHS * DIM, B), B, 0, stream>>>(rvec, N_GRAPHS * DIM);
  mpnn_e<<<gb(N_NODES, B), B, 0, stream>>>(s32_cur, q0, batch, e_buf, emax);
  mpnn_ee<<<gb(N_NODES, B), B, 0, stream>>>(e_buf, emax, batch, ee_buf, denom);
  mpnn_rvec<<<gb(N_NODES * DIM, B), B, 0, stream>>>(s32_cur, ee_buf, denom, batch, rvec);
  mpnn_out<<<gb(N_GRAPHS * 2 * DIM, B), B, 0, stream>>>(q0, rvec, (float*)d_out);
}
